// UmapLoss_refine_conf_13915694039295
// MI455X (gfx1250) — compile-verified, matrix operand order checked by reading, not run on hardware
//
#include <hip/hip_runtime.h>
#include <math.h>

#define BATCH 65536
#define CCLS  100
#define RATE  5
#define NEG   (BATCH * RATE)   // 327680
#define ECNT  2048
#define GNN   16
#define GPP   5
#define CEPS  1e-4f

typedef __attribute__((ext_vector_type(2)))  float    v2f;
typedef __attribute__((ext_vector_type(8)))  float    v8f;
typedef __attribute__((ext_vector_type(16))) _Float16 v16h;
typedef __attribute__((ext_vector_type(2)))  int      v2i;

typedef __attribute__((address_space(1))) v2i* gptr_v2i;
typedef __attribute__((address_space(3))) v2i* lptr_v2i;

#define HAS_ASYNC_LDS __has_builtin(__builtin_amdgcn_global_load_async_to_lds_b64)

// ---------------------------------------------------------------------------
// Kernel 1: per-row argmax + max-softmax confidence for pred_to / pred_from.
// 8 waves/block, 16 rows/wave (128 rows/block), gridDim=(512,2).
// Staging: async global->LDS (ASYNCcnt path) when available.
// Row sums of exp(x-max) computed with WMMA (B = ones) over 25 K=4 chunks.
// ---------------------------------------------------------------------------
__global__ __launch_bounds__(256) void k1_rowstats(
    const float* __restrict__ predA, const float* __restrict__ predB,
    float* __restrict__ cA, float* __restrict__ cB,
    int* __restrict__ gA, int* __restrict__ gB)
{
    const float* pred = blockIdx.y ? predB : predA;
    float* conf       = blockIdx.y ? cB : cA;
    int*   amax       = blockIdx.y ? gB : gA;

    __shared__ float tile[8][16 * 104];   // 16 rows x 100 cols, stride 104 (53 KB)
    const int wave = threadIdx.x >> 5;
    const int lane = threadIdx.x & 31;
    const int rowBase = (blockIdx.x * 8 + wave) * 16;

    // Coalesced staging: 16 contiguous rows = 1600 floats = 800 float2 / wave.
    const float2* gsrc = (const float2*)(pred + (size_t)rowBase * CCLS);
#if HAS_ASYNC_LDS
#pragma unroll
    for (int j = 0; j < 25; ++j) {
        int flat = j * 32 + lane;          // float2 index in [0,800)
        int e0 = flat * 2;
        int r = e0 / 100;
        int c = e0 - r * 100;              // even -> 8B aligned
        __builtin_amdgcn_global_load_async_to_lds_b64(
            (gptr_v2i)(void*)(gsrc + flat),
            (lptr_v2i)(void*)&tile[wave][r * 104 + c],
            0, 0);
    }
    asm volatile("s_wait_asynccnt 0x0" ::: "memory");
#else
#pragma unroll
    for (int j = 0; j < 25; ++j) {
        int flat = j * 32 + lane;
        float2 v = gsrc[flat];
        int e0 = flat * 2;
        int r = e0 / 100;
        int c = e0 - r * 100;
        *(float2*)&tile[wave][r * 104 + c] = v;
    }
#endif
    __syncthreads();

    // WMMA A layout: lane m holds row m cols {4j,4j+1}; lane m+16 holds {4j+2,4j+3}
    const int m = lane & 15, h = lane >> 4;
    float2 vals[25];
#pragma unroll
    for (int j = 0; j < 25; ++j)
        vals[j] = *(const float2*)&tile[wave][m * 104 + 4 * j + 2 * h];

    // per-lane max/argmax over its 50 columns (ascending col order -> first-max)
    float mx = -3.4e38f; int ai = 0;
#pragma unroll
    for (int j = 0; j < 25; ++j) {
        int c0 = 4 * j + 2 * h;
        if (vals[j].x > mx) { mx = vals[j].x; ai = c0; }
        if (vals[j].y > mx) { mx = vals[j].y; ai = c0 + 1; }
    }
    // combine lane pair (m, m+16); tie-break on smaller column index
    float omx = __shfl_xor(mx, 16, 32);
    int   oai = __shfl_xor(ai, 16, 32);
    if (omx > mx || (omx == mx && oai < ai)) { mx = omx; ai = oai; }

    // Row sums of exp(x - max) via WMMA with B = ones.
    v8f acc = {0.f, 0.f, 0.f, 0.f, 0.f, 0.f, 0.f, 0.f};
#if __has_builtin(__builtin_amdgcn_wmma_f32_16x16x4_f32)
    v2f ones; ones.x = 1.0f; ones.y = 1.0f;
#pragma unroll
    for (int j = 0; j < 25; ++j) {
        v2f a; a.x = __expf(vals[j].x - mx); a.y = __expf(vals[j].y - mx);
        acc = __builtin_amdgcn_wmma_f32_16x16x4_f32(
            false, a, false, ones, (short)0, acc, false, false);
    }
#else
    // Fallback: codegen-confirmed f16 WMMA. With B = all-ones the A slot
    // layout is irrelevant for a row sum; pad 50 values -> 64 slots with 0.
    v16h ones16;
#pragma unroll
    for (int e = 0; e < 16; ++e) ones16[e] = (_Float16)1.0f;
#pragma unroll
    for (int t = 0; t < 4; ++t) {
        v16h a16;
#pragma unroll
        for (int e = 0; e < 16; ++e) {
            int k = t * 16 + e;
            float v = 0.0f;
            if (k < 50) {
                float x = (k & 1) ? vals[k >> 1].y : vals[k >> 1].x;
                v = __expf(x - mx);
            }
            a16[e] = (_Float16)v;
        }
        acc = __builtin_amdgcn_wmma_f32_16x16x32_f16(
            false, a16, false, ones16, (short)0, acc, false, false);
    }
#endif

    // D layout: VGPR r = row r (lanes 0-15) / row 8+r (lanes 16-31); N = lane.
    if (lane < 16) amax[rowBase + lane] = ai;
    if (lane == 0) {
#pragma unroll
        for (int r = 0; r < 8; ++r) conf[rowBase + r] = 1.0f / acc[r];
    } else if (lane == 16) {
#pragma unroll
        for (int r = 0; r < 8; ++r) conf[rowBase + 8 + r] = 1.0f / acc[r];
    }
}

// ---------------------------------------------------------------------------
// wave32 deterministic reduction helper
// ---------------------------------------------------------------------------
__device__ __forceinline__ float wred(float v)
{
#pragma unroll
    for (int o = 16; o > 0; o >>= 1) v += __shfl_xor(v, o, 32);
    return v;
}

// ---------------------------------------------------------------------------
// Kernel 2: negative-sampling pass. 1280 blocks x 256 = 327680.
// ---------------------------------------------------------------------------
__global__ __launch_bounds__(256) void k2_neg(
    const int* __restrict__ perm,
    const float2* __restrict__ embTo, const float2* __restrict__ embFrom,
    const float* __restrict__ confTo, const float* __restrict__ confFrom,
    const int* __restrict__ argTo, const int* __restrict__ argFrom,
    float* __restrict__ pCnt, float* __restrict__ pDist, float* __restrict__ pCe)
{
    const int i = blockIdx.x * 256 + threadIdx.x;
    const int to = i / RATE;
    const int from = perm[i] / RATE;

    bool same = (argFrom[from] == argTo[to]) && (confFrom[from] == confTo[to]);
    float mneg = same ? 0.0f : 1.0f;

    float2 a = embTo[to], b = embFrom[from];
    float dx = a.x - b.x, dy = a.y - b.y;
    float d2 = dx * dx + dy * dy;
    float d  = sqrtf(d2);
    float omq = d2 / (1.0f + d2);                 // 1 - q(d)
    float ce = -__logf(fmaxf(omq, CEPS));         // _ce_repulse(0, q)

    float sc = wred(mneg);
    float sd = wred(mneg * d);
    float se = wred(mneg * ce);

    __shared__ float red[3][8];
    const int wave = threadIdx.x >> 5, lane = threadIdx.x & 31;
    if (lane == 0) { red[0][wave] = sc; red[1][wave] = sd; red[2][wave] = se; }
    __syncthreads();
    if (threadIdx.x == 0) {
        float t0 = 0, t1 = 0, t2 = 0;
        for (int w = 0; w < 8; ++w) { t0 += red[0][w]; t1 += red[1][w]; t2 += red[2][w]; }
        pCnt[blockIdx.x] = t0; pDist[blockIdx.x] = t1; pCe[blockIdx.x] = t2;
    }
}

// ---------------------------------------------------------------------------
// Kernel 3: per-edge pass: pos dist, searchsorted hits, grid CE terms.
// 256 blocks x 256 = 65536.
// ---------------------------------------------------------------------------
__global__ __launch_bounds__(256) void k3_edge(
    const int* __restrict__ eTo, const int* __restrict__ eFrom,
    const float2* __restrict__ embTo, const float2* __restrict__ embFrom,
    const float* __restrict__ probs, const int* __restrict__ errConf,
    const float2* __restrict__ negGrid, const float2* __restrict__ posGrid,
    const float* __restrict__ distList,
    float* __restrict__ pCe, float* __restrict__ pW,
    float* __restrict__ pM, float* __restrict__ pPos)
{
    __shared__ int ec[ECNT];
    for (int t = threadIdx.x; t < ECNT; t += 256) ec[t] = errConf[t];
    __syncthreads();

    const int b = blockIdx.x * 256 + threadIdx.x;

    // searchsorted (side=left) + clip + hit test, against LDS-resident table
    auto search = [&](int v, float& hit) -> int {
        int lo = 0, hi = ECNT;
        while (lo < hi) { int mid = (lo + hi) >> 1; if (ec[mid] < v) lo = mid + 1; else hi = mid; }
        int pos = lo > (ECNT - 1) ? (ECNT - 1) : lo;
        hit = (ec[pos] == v) ? 1.0f : 0.0f;
        return pos;
    };
    float mt, mf;
    const int gto = search(eTo[b], mt);
    const int gfr = search(eFrom[b], mf);

    const float2 et = embTo[b], ef = embFrom[b];
    const float dx = et.x - ef.x, dy = et.y - ef.y;
    const float pd2 = dx * dx + dy * dy;
    const float pd = sqrtf(pd2);
    const float p = probs[b];

    // ce_pos = attract(p,q) + repulse(p,q)
    float q = 1.0f / (1.0f + pd2);
    float s = -p * __logf(fmaxf(q, CEPS))
              - (1.0f - p) * __logf(fmaxf(pd2 / (1.0f + pd2), CEPS));

    const float wpt = mt * (distList[gto] < 0.5f ? 1.0f : 0.0f);
    const float wpf = mf * (distList[gfr] < 1.0f ? 1.0f : 0.0f);

    // attract terms: embedding_from vs pos_grid[g_to] and pos_grid[g_from]
    float sa_t = 0.f, sa_f = 0.f;
#pragma unroll
    for (int k = 0; k < GPP; ++k) {
        float2 g = posGrid[gto * GPP + k];
        float ax = ef.x - g.x, ay = ef.y - g.y;
        float d2 = ax * ax + ay * ay;
        sa_t += -__logf(fmaxf(1.0f / (1.0f + d2), CEPS));
        g = posGrid[gfr * GPP + k];
        ax = ef.x - g.x; ay = ef.y - g.y;
        d2 = ax * ax + ay * ay;
        sa_f += -__logf(fmaxf(1.0f / (1.0f + d2), CEPS));
    }
    s += wpt * sa_t + wpf * sa_f;

    // repulse terms: embedding_to vs neg_grid[g_to]; embedding_from vs neg_grid[g_from]
    float sr_t = 0.f, sr_f = 0.f;
#pragma unroll
    for (int k = 0; k < GNN; ++k) {
        float2 g = negGrid[gto * GNN + k];
        float ax = et.x - g.x, ay = et.y - g.y;
        float d2 = ax * ax + ay * ay;
        sr_t += -__logf(fmaxf(d2 / (1.0f + d2), CEPS));
        g = negGrid[gfr * GNN + k];
        ax = ef.x - g.x; ay = ef.y - g.y;
        d2 = ax * ax + ay * ay;
        sr_f += -__logf(fmaxf(d2 / (1.0f + d2), CEPS));
    }
    s += mt * sr_t + mf * sr_f;

    float sce = wred(s);
    float sw  = wred(wpt + wpf);
    float sm  = wred(mt + mf);
    float sp  = wred(pd);

    __shared__ float red[4][8];
    const int wave = threadIdx.x >> 5, lane = threadIdx.x & 31;
    if (lane == 0) { red[0][wave] = sce; red[1][wave] = sw; red[2][wave] = sm; red[3][wave] = sp; }
    __syncthreads();
    if (threadIdx.x == 0) {
        float t0 = 0, t1 = 0, t2 = 0, t3 = 0;
        for (int w = 0; w < 8; ++w) { t0 += red[0][w]; t1 += red[1][w]; t2 += red[2][w]; t3 += red[3][w]; }
        pCe[blockIdx.x] = t0; pW[blockIdx.x] = t1; pM[blockIdx.x] = t2; pPos[blockIdx.x] = t3;
    }
}

// ---------------------------------------------------------------------------
// Kernel 4: reduce partials -> umap_l, pos_mean, neg_mean. Single block.
// ---------------------------------------------------------------------------
__global__ __launch_bounds__(256) void k4_reduceA(
    const float* __restrict__ pCnt, const float* __restrict__ pDist,
    const float* __restrict__ pCe2, const float* __restrict__ pCe3,
    const float* __restrict__ pW, const float* __restrict__ pM,
    const float* __restrict__ pPos, float* __restrict__ scal)
{
    const int t = threadIdx.x;
    float a0 = 0, a1 = 0, a2 = 0;
    for (int i = t; i < 1280; i += 256) { a0 += pCnt[i]; a1 += pDist[i]; a2 += pCe2[i]; }
    float a3 = pCe3[t], a4 = pW[t], a5 = pM[t], a6 = pPos[t];

    a0 = wred(a0); a1 = wred(a1); a2 = wred(a2); a3 = wred(a3);
    a4 = wred(a4); a5 = wred(a5); a6 = wred(a6);

    __shared__ float red[7][8];
    const int wave = threadIdx.x >> 5, lane = threadIdx.x & 31;
    if (lane == 0) {
        red[0][wave] = a0; red[1][wave] = a1; red[2][wave] = a2; red[3][wave] = a3;
        red[4][wave] = a4; red[5][wave] = a5; red[6][wave] = a6;
    }
    __syncthreads();
    if (t == 0) {
        float s[7];
        for (int j = 0; j < 7; ++j) {
            float acc = 0;
            for (int w = 0; w < 8; ++w) acc += red[j][w];
            s[j] = acc;
        }
        const float negcnt = s[0], negd = s[1], negce = s[2];
        const float ce3 = s[3], wsum = s[4], msum = s[5], possum = s[6];
        float pos_mean = possum / (float)BATCH;
        float neg_mean = negd / fmaxf(negcnt, 1.0f);
        float sum_ce = ce3 + negce;
        float total = (float)BATCH + (float)GPP * wsum + negcnt + (float)GNN * msum;
        float umap = sum_ce / fmaxf(total, 1.0f);
        if (umap != umap) umap = 0.0f;
        scal[0] = umap; scal[1] = pos_mean; scal[2] = neg_mean;
    }
}

// ---------------------------------------------------------------------------
// Kernel 5: margin loss partials. 256 blocks x 256.
// ---------------------------------------------------------------------------
__global__ __launch_bounds__(256) void k5_margin(
    const float2* __restrict__ embTo, const float2* __restrict__ embFrom,
    const float* __restrict__ probs,
    const int* __restrict__ argTo, const int* __restrict__ argFrom,
    const float* __restrict__ scal, float* __restrict__ pOut)
{
    const int b = blockIdx.x * 256 + threadIdx.x;
    const float pos_mean = scal[1], neg_mean = scal[2];

    float2 et = embTo[b], ef = embFrom[b];
    float dx = et.x - ef.x, dy = et.y - ef.y;
    float pd = sqrtf(dx * dx + dy * dy);
    float isSame = (argTo[b] == argFrom[b]) ? 1.0f : 0.0f;
    float bm = pos_mean + (neg_mean - pos_mean) * (1.0f - probs[b]);
    float v = fmaxf((1.0f - isSame) * bm - pd, 0.0f);

    v = wred(v);
    __shared__ float red[8];
    const int wave = threadIdx.x >> 5, lane = threadIdx.x & 31;
    if (lane == 0) red[wave] = v;
    __syncthreads();
    if (threadIdx.x == 0) {
        float t0 = 0;
        for (int w = 0; w < 8; ++w) t0 += red[w];
        pOut[blockIdx.x] = t0;
    }
}

// ---------------------------------------------------------------------------
// Kernel 6: final reduce -> (umap_l, margin_loss, sum). Single block of 256.
// ---------------------------------------------------------------------------
__global__ __launch_bounds__(256) void k6_final(
    const float* __restrict__ pOut, const float* __restrict__ scal,
    float* __restrict__ out)
{
    float v = pOut[threadIdx.x];
    v = wred(v);
    __shared__ float red[8];
    const int wave = threadIdx.x >> 5, lane = threadIdx.x & 31;
    if (lane == 0) red[wave] = v;
    __syncthreads();
    if (threadIdx.x == 0) {
        float t0 = 0;
        for (int w = 0; w < 8; ++w) t0 += red[w];
        float ml = t0 / (float)BATCH;
        if (ml != ml) ml = 0.0f;
        float u = scal[0];
        out[0] = u; out[1] = ml; out[2] = u + ml;
    }
}

// ---------------------------------------------------------------------------
extern "C" void kernel_launch(void* const* d_in, const int* in_sizes, int n_in,
                              void* d_out, int out_size, void* d_ws, size_t ws_size,
                              hipStream_t stream)
{
    (void)in_sizes; (void)n_in; (void)out_size; (void)ws_size;

    const int*    edgeTo   = (const int*)d_in[0];
    const int*    edgeFrom = (const int*)d_in[1];
    const float2* embTo    = (const float2*)d_in[2];
    const float2* embFrom  = (const float2*)d_in[3];
    const float*  probs    = (const float*)d_in[4];
    const float*  predTo   = (const float*)d_in[5];
    const float*  predFrom = (const float*)d_in[6];
    const int*    perm     = (const int*)d_in[7];
    const int*    errConf  = (const int*)d_in[8];
    const float2* negGrid  = (const float2*)d_in[9];
    const float2* posGrid  = (const float2*)d_in[10];
    const float*  distList = (const float*)d_in[11];
    float*        out      = (float*)d_out;

    // workspace layout (floats)
    float* ws        = (float*)d_ws;
    float* conf_to   = ws;                     // 65536
    float* conf_from = ws + BATCH;             // 65536
    int*   arg_to    = (int*)(ws + 2 * BATCH); // 65536
    int*   arg_from  = (int*)(ws + 3 * BATCH); // 65536
    float* p2cnt     = ws + 4 * BATCH;         // 1280
    float* p2d       = p2cnt + 1280;
    float* p2ce      = p2d + 1280;
    float* p3ce      = p2ce + 1280;            // 256 each
    float* p3w       = p3ce + 256;
    float* p3m       = p3w + 256;
    float* p3pos     = p3m + 256;
    float* scal      = p3pos + 256;            // 3
    float* p5        = scal + 3;               // 256

    k1_rowstats<<<dim3(512, 2), 256, 0, stream>>>(
        predTo, predFrom, conf_to, conf_from, arg_to, arg_from);

    k2_neg<<<NEG / 256, 256, 0, stream>>>(
        perm, embTo, embFrom, conf_to, conf_from, arg_to, arg_from,
        p2cnt, p2d, p2ce);

    k3_edge<<<BATCH / 256, 256, 0, stream>>>(
        edgeTo, edgeFrom, embTo, embFrom, probs, errConf,
        negGrid, posGrid, distList, p3ce, p3w, p3m, p3pos);

    k4_reduceA<<<1, 256, 0, stream>>>(
        p2cnt, p2d, p2ce, p3ce, p3w, p3m, p3pos, scal);

    k5_margin<<<BATCH / 256, 256, 0, stream>>>(
        embTo, embFrom, probs, arg_to, arg_from, scal, p5);

    k6_final<<<1, 256, 0, stream>>>(p5, scal, out);
}
